// BraggNN_6717328850963
// MI455X (gfx1250) — matrix operand order repaired and checked by reading, not compiled, not run on hardware
//
#include <hip/hip_runtime.h>

typedef _Float16 half_t;
typedef __attribute__((ext_vector_type(8)))  _Float16 v8h;
typedef __attribute__((ext_vector_type(16))) _Float16 v16h;
typedef __attribute__((ext_vector_type(8)))  float    v8f;

#define THREADS 128

// ---------------- packed f16 weight layout in d_ws (element offsets) --------
// wt/wp/wg: [32][64] row-major. wo: [64][32] row-major.
// w2r: [32][576], K reordered tap-major: k = (ky*3+kx)*64 + c
// w3r: [8][288],  K reordered tap-major: k = (ky*3+kx)*32 + c
#define WOFF_WT 0
#define WOFF_WP 2048
#define WOFF_WG 4096
#define WOFF_WO 6144
#define WOFF_W2 8192
#define WOFF_W3 26624
#define WTOTAL  28928

// ---------------- LDS layout (byte offsets, all 16B-aligned) ----------------
#define OFF_A1F  0                        // float [64*81]   conv1 out (residual)
#define OFF_A1T  (OFF_A1F + 64*81*4)      // half  [96][64]  conv1 out, spatial-major
#define OFF_THT  (OFF_A1T + 96*64*2)      // half  [96][32]  theta^T  (s-major)
#define OFF_PHT  (OFF_THT + 96*32*2)      // half  [96][32]  phi^T    (t-major)
#define OFF_GT   (OFF_PHT + 96*32*2)      // half  [32][96]  g        (i-major)
#define OFF_BIG  (OFF_GT  + 32*96*2)      // float [96][96]  attn logits
#define OFF_ATTN (OFF_BIG + 96*96*4)      // half  [96][96]  attn (row-major)
#define OFF_YH   (OFF_ATTN + 96*96*2)     // half  [96][32]  y (s-major)
#define OFF_A2T  (OFF_YH + 96*32*2)       // half  [81][64]  stage-2 act, spatial-major
#define OFF_A3T  (OFF_A2T + 81*64*2)      // half  [49][32]  stage-3 act, spatial-major
#define OFF_CD   (OFF_A3T + 49*32*2)      // float [16*32]   conv3 D
#define OFF_XIN  (OFF_CD + 16*32*4)       // float [124]
#define OFF_FC   (OFF_XIN + 124*4)        // float [200]
#define OFF_FC2  (OFF_FC + 200*4)         // float [64]
#define SMEM_BYTES (OFF_FC2 + 64*4)

__device__ __forceinline__ float lrelu(float v) { return v > 0.f ? v : 0.01f * v; }

__device__ __forceinline__ v16h frag_cat(v8h c0, v8h c1) {
  return __builtin_shufflevector(c0, c1, 0, 1, 2, 3, 4, 5, 6, 7,
                                 8, 9, 10, 11, 12, 13, 14, 15);
}

// Generic 16x16-tile GEMM over the CDNA5 f16 fragment layout.
// pa(m,k0) -> const half_t* : pointer to 8 contiguous halves = A(m, k0..k0+7)
// pb(n,k0) -> const half_t* : pointer to 8 contiguous halves = B(k0..k0+7, n)
// fd(m,n,val)               : epilogue store of D element.
// A and B chunks both sit at k0 = kb+8*hi / kb+16+8*hi
// (ISA 7.12.2 interleaved half layout for both operands). Tile loop is
// wave-uniform so EXEC is all-ones at each WMMA.
template <typename PA, typename PB, typename FD>
__device__ __forceinline__ void wmma_gemm(PA pa, PB pb, FD fd,
                                          int Mt, int Nt, int Kt) {
  const int wave = (int)(threadIdx.x >> 5);
  const int nw   = (int)(blockDim.x >> 5);
  const int lane = (int)(threadIdx.x & 31);
  const int lo = lane & 15;
  const int hi = lane >> 4;
  for (int t = wave; t < Mt * Nt; t += nw) {
    const int mi = t / Nt, ni = t - mi * Nt;
    const int am = mi * 16 + lo;
    const int bn = ni * 16 + lo;
    v8f acc = {0.f, 0.f, 0.f, 0.f, 0.f, 0.f, 0.f, 0.f};
    for (int kk = 0; kk < Kt; ++kk) {
      const int kb = kk * 32;
      v8h a0 = *(const v8h*)pa(am, kb + 8 * hi);
      v8h a1 = *(const v8h*)pa(am, kb + 16 + 8 * hi);
      v8h b0 = *(const v8h*)pb(bn, kb + 8 * hi);
      v8h b1 = *(const v8h*)pb(bn, kb + 16 + 8 * hi);
      acc = __builtin_amdgcn_wmma_f32_16x16x32_f16(
          false, frag_cat(a0, a1), false, frag_cat(b0, b1),
          (short)0, acc, false, false);
    }
    const int n  = ni * 16 + lo;
    const int m0 = mi * 16 + hi * 8;
#pragma unroll
    for (int r = 0; r < 8; ++r) fd(m0 + r, n, acc[r]);
  }
}

// ------------- pre-kernel: convert/permute weights f32 -> f16 into d_ws -----
__global__ void cvt_weights(const float* __restrict__ wt, const float* __restrict__ wp,
                            const float* __restrict__ wg, const float* __restrict__ wo,
                            const float* __restrict__ w2, const float* __restrict__ w3,
                            half_t* __restrict__ dst) {
  int i = blockIdx.x * 256 + threadIdx.x;
  if (i >= WTOTAL) return;
  float v;
  if      (i < WOFF_WP) v = wt[i - WOFF_WT];
  else if (i < WOFF_WG) v = wp[i - WOFF_WP];
  else if (i < WOFF_WO) v = wg[i - WOFF_WG];
  else if (i < WOFF_W2) v = wo[i - WOFF_WO];
  else if (i < WOFF_W3) {             // w2 [32][64][9] -> [32][tap*64+c]
    int j = i - WOFF_W2;
    int m = j / 576, r = j - m * 576;
    int tap = r >> 6, c = r & 63;
    v = w2[(m * 64 + c) * 9 + tap];
  } else {                            // w3 [8][32][9] -> [8][tap*32+c]
    int j = i - WOFF_W3;
    int m = j / 288, r = j - m * 288;
    int tap = r >> 5, c = r & 31;
    v = w3[(m * 32 + c) * 9 + tap];
  }
  dst[i] = (half_t)v;
}

// ------------- fused per-image kernel ---------------------------------------
__global__ __launch_bounds__(THREADS) void braggnn_fused(
    const float* __restrict__ x,  const float* __restrict__ w1, const float* __restrict__ b1,
    const float* __restrict__ bt, const float* __restrict__ bp, const float* __restrict__ bg,
    const float* __restrict__ bo, const float* __restrict__ b2, const float* __restrict__ b3,
    const float* __restrict__ fw1, const float* __restrict__ fb1,
    const float* __restrict__ fw2, const float* __restrict__ fb2,
    const float* __restrict__ fw3, const float* __restrict__ fb3,
    const float* __restrict__ fw4, const float* __restrict__ fb4,
    const float* __restrict__ fwo, const float* __restrict__ fbo,
    const half_t* __restrict__ wh, float* __restrict__ out) {
  extern __shared__ unsigned char smem[];
  float*  a1f  = (float*)(smem + OFF_A1F);
  half_t* a1t  = (half_t*)(smem + OFF_A1T);
  half_t* th_t = (half_t*)(smem + OFF_THT);
  half_t* ph_t = (half_t*)(smem + OFF_PHT);
  half_t* g_t  = (half_t*)(smem + OFF_GT);
  float*  big  = (float*)(smem + OFF_BIG);
  half_t* attn = (half_t*)(smem + OFF_ATTN);
  half_t* yh   = (half_t*)(smem + OFF_YH);
  half_t* a2t  = (half_t*)(smem + OFF_A2T);
  half_t* a3t  = (half_t*)(smem + OFF_A3T);
  float*  cd   = (float*)(smem + OFF_CD);
  float*  xin  = (float*)(smem + OFF_XIN);
  float*  sfc  = (float*)(smem + OFF_FC);
  float*  sfc2 = (float*)(smem + OFF_FC2);

  const int tid = (int)threadIdx.x;
  const int img = (int)blockIdx.x;

  const half_t* wth = wh + WOFF_WT;
  const half_t* wph = wh + WOFF_WP;
  const half_t* wgh = wh + WOFF_WG;
  const half_t* woh = wh + WOFF_WO;
  const half_t* w2h = wh + WOFF_W2;
  const half_t* w3h = wh + WOFF_W3;

  // ---- stage 0: stage input image in LDS -----------------------------------
  const float* xg = x + (size_t)img * 121;
  for (int i = tid; i < 121; i += THREADS) xin[i] = xg[i];
  __syncthreads();

  // ---- stage 1: conv1 1->64, 3x3 valid, scalar f32 (tiny K=9) --------------
  for (int i = tid; i < 64 * 81; i += THREADS) {
    int c = i / 81, s = i - c * 81;
    int oy = s / 9, ox = s - oy * 9;
    float acc = b1[c];
#pragma unroll
    for (int ky = 0; ky < 3; ++ky)
#pragma unroll
      for (int kx = 0; kx < 3; ++kx)
        acc += xin[(oy + ky) * 11 + (ox + kx)] * w1[c * 9 + ky * 3 + kx];
    a1f[i] = acc;
    a1t[s * 64 + c] = (half_t)acc;   // spatial-major, K(=c)-contiguous
  }
  for (int i = tid; i < 15 * 64; i += THREADS) {
    int s = 81 + i / 64, c = i & 63;
    a1t[s * 64 + c] = (half_t)0.f;   // zero N-padding rows
  }
  __syncthreads();

  // ---- stage 2: theta/phi/g = W[32,64] x a1(64,96) (WMMA), bias fused ------
  // theta/phi stored transposed [s][i]; g stored [i][t].
  {
    auto pb = [&](int n, int k0) -> const half_t* { return a1t + n * 64 + k0; };
    for (int q = 0; q < 2; ++q) {
      const half_t* W = q == 0 ? wth : wph;
      const float* bias = q == 0 ? bt : bp;
      half_t* dstT = q == 0 ? th_t : ph_t;
      wmma_gemm(
          [&](int m, int k0) -> const half_t* { return W + m * 64 + k0; },
          pb,
          [&](int m, int n, float v) { dstT[n * 32 + m] = (half_t)(v + bias[m]); },
          2, 6, 2);
    }
    wmma_gemm(
        [&](int m, int k0) -> const half_t* { return wgh + m * 64 + k0; },
        pb,
        [&](int m, int n, float v) { g_t[m * 96 + n] = (half_t)(v + bg[m]); },
        2, 6, 2);
  }
  __syncthreads();

  // ---- stage 3: logits[96,96] = theta^T x phi (WMMA, K=32) -----------------
  wmma_gemm(
      [&](int m, int k0) -> const half_t* { return th_t + m * 32 + k0; },
      [&](int n, int k0) -> const half_t* { return ph_t + n * 32 + k0; },
      [&](int m, int n, float v) { big[m * 96 + n] = v; },
      6, 6, 1);
  __syncthreads();

  // ---- stage 4: row softmax over t<81; write f16 attn, zero padding --------
  for (int srow = tid; srow < 96; srow += THREADS) {
    if (srow < 81) {
      const float* row = big + srow * 96;
      float mx = -1e30f;
      for (int t = 0; t < 81; ++t) mx = fmaxf(mx, row[t]);
      float sum = 0.f;
      for (int t = 0; t < 81; ++t) sum += __expf(row[t] - mx);
      float inv = 1.f / sum;
      for (int t = 0; t < 81; ++t)
        attn[srow * 96 + t] = (half_t)(__expf(row[t] - mx) * inv);
      for (int t = 81; t < 96; ++t) attn[srow * 96 + t] = (half_t)0.f;
    } else {
      for (int t = 0; t < 96; ++t) attn[srow * 96 + t] = (half_t)0.f;
    }
  }
  __syncthreads();

  // ---- stage 5: y[96,32] = attn[96,96] x g^T (WMMA, K=96) ------------------
  wmma_gemm(
      [&](int m, int k0) -> const half_t* { return attn + m * 96 + k0; },
      [&](int n, int k0) -> const half_t* { return g_t + n * 96 + k0; },
      [&](int m, int n, float v) { yh[m * 32 + n] = (half_t)v; },
      6, 2, 3);
  __syncthreads();

  // ---- stage 6: wo[64,32] x y^T; + bo + residual, LReLU -> a2t[s][c] -------
  wmma_gemm(
      [&](int m, int k0) -> const half_t* { return woh + m * 32 + k0; },
      [&](int n, int k0) -> const half_t* { return yh + n * 32 + k0; },
      [&](int m, int n, float v) {
        if (n < 81) {
          float r = v + bo[m] + a1f[m * 81 + n];
          a2t[n * 64 + m] = (half_t)lrelu(r);
        }
      },
      4, 6, 1);
  __syncthreads();

  // ---- stage 7: conv2 64->32, implicit GEMM, K tap-major (=9 taps x 64c) ---
  wmma_gemm(
      [&](int m, int k0) -> const half_t* { return w2h + m * 576 + k0; },
      [&](int n, int k0) -> const half_t* {
        int nn = n < 49 ? n : 48;        // clamp N padding into valid memory
        int oy = nn / 7, ox = nn - oy * 7;
        int tap = k0 >> 6, c0 = k0 & 63; // 8-half chunk never crosses a c-run
        int ky = tap / 3, kx = tap - ky * 3;
        return a2t + ((oy + ky) * 9 + (ox + kx)) * 64 + c0;
      },
      [&](int m, int n, float v) {
        if (n < 49) a3t[n * 32 + m] = (half_t)lrelu(v + b2[m]);
      },
      2, 4, 18);
  __syncthreads();

  // ---- stage 8: conv3 32->8, implicit GEMM, K tap-major (=9 taps x 32c) ----
  wmma_gemm(
      [&](int m, int k0) -> const half_t* {
        int mm = m < 8 ? m : 7;          // clamp M padding; D rows 8..15 unused
        return w3h + mm * 288 + k0;
      },
      [&](int n, int k0) -> const half_t* {
        int nn = n < 25 ? n : 24;
        int oy = nn / 5, ox = nn - oy * 5;
        int tap = k0 >> 5, c0 = k0 & 31;
        int ky = tap / 3, kx = tap - ky * 3;
        return a3t + ((oy + ky) * 7 + (ox + kx)) * 32 + c0;
      },
      [&](int m, int n, float v) { cd[m * 32 + n] = v; },
      1, 2, 9);
  __syncthreads();

  // ---- stage 9: flatten + bias + LReLU -> [200] ----------------------------
  for (int i = tid; i < 200; i += THREADS) {
    int c = i / 25, p = i - c * 25;
    sfc[i] = lrelu(cd[c * 32 + p] + b3[c]);
  }
  __syncthreads();

  // ---- stage 10: FC chain (tiny, scalar f32; weights L2-resident) ----------
  if (tid < 64) {
    float acc = fb1[tid];
    for (int j = 0; j < 200; ++j) acc += fw1[tid * 200 + j] * sfc[j];
    sfc2[tid] = lrelu(acc);
  }
  __syncthreads();
  if (tid < 32) {
    float acc = fb2[tid];
    for (int j = 0; j < 64; ++j) acc += fw2[tid * 64 + j] * sfc2[j];
    sfc[tid] = lrelu(acc);
  }
  __syncthreads();
  if (tid < 16) {
    float acc = fb3[tid];
    for (int j = 0; j < 32; ++j) acc += fw3[tid * 32 + j] * sfc[j];
    sfc2[tid] = lrelu(acc);
  }
  __syncthreads();
  if (tid < 8) {
    float acc = fb4[tid];
    for (int j = 0; j < 16; ++j) acc += fw4[tid * 16 + j] * sfc2[j];
    sfc[tid] = lrelu(acc);
  }
  __syncthreads();
  if (tid < 2) {
    float acc = fbo[tid];
    for (int j = 0; j < 8; ++j) acc += fwo[tid * 8 + j] * sfc[j];
    out[(size_t)img * 2 + tid] = acc;
  }
}

extern "C" void kernel_launch(void* const* d_in, const int* in_sizes, int n_in,
                              void* d_out, int out_size, void* d_ws, size_t ws_size,
                              hipStream_t stream) {
  const float* x   = (const float*)d_in[0];
  const float* w1  = (const float*)d_in[1];
  const float* b1  = (const float*)d_in[2];
  const float* wt  = (const float*)d_in[3];
  const float* bt  = (const float*)d_in[4];
  const float* wp  = (const float*)d_in[5];
  const float* bp  = (const float*)d_in[6];
  const float* wg  = (const float*)d_in[7];
  const float* bg  = (const float*)d_in[8];
  const float* wo  = (const float*)d_in[9];
  const float* bo  = (const float*)d_in[10];
  const float* w2  = (const float*)d_in[11];
  const float* b2  = (const float*)d_in[12];
  const float* w3  = (const float*)d_in[13];
  const float* b3  = (const float*)d_in[14];
  const float* fw1 = (const float*)d_in[15];
  const float* fb1 = (const float*)d_in[16];
  const float* fw2 = (const float*)d_in[17];
  const float* fb2 = (const float*)d_in[18];
  const float* fw3 = (const float*)d_in[19];
  const float* fb3 = (const float*)d_in[20];
  const float* fw4 = (const float*)d_in[21];
  const float* fb4 = (const float*)d_in[22];
  const float* fwo = (const float*)d_in[23];
  const float* fbo = (const float*)d_in[24];

  half_t* wh = (half_t*)d_ws;
  const int nimg = in_sizes[0] / 121;

  cvt_weights<<<(WTOTAL + 255) / 256, 256, 0, stream>>>(wt, wp, wg, wo, w2, w3, wh);

  braggnn_fused<<<nimg, THREADS, SMEM_BYTES, stream>>>(
      x, w1, b1, bt, bp, bg, bo, b2, b3,
      fw1, fb1, fw2, fb2, fw3, fb3, fw4, fb4, fwo, fbo,
      wh, (float*)d_out);
}
